// Block_24111946399747
// MI455X (gfx1250) — matrix operand order repaired and checked by reading, not compiled, not run on hardware
//
#include <hip/hip_runtime.h>

// ---------------------------------------------------------------------------
// Types for CDNA5 WMMA (wave32): v16bf A/B fragments, v8f C/D accumulators.
// ---------------------------------------------------------------------------
typedef __bf16 bf16;
typedef unsigned int u32;
typedef __attribute__((ext_vector_type(16))) __bf16 v16bf;
typedef __attribute__((ext_vector_type(8)))  __bf16 v8bf;
typedef __attribute__((ext_vector_type(8)))  float  v8f;
typedef __attribute__((ext_vector_type(4)))  unsigned int v4u;
typedef __attribute__((ext_vector_type(8)))  int v8i;
typedef __attribute__((ext_vector_type(4)))  int v4i;

__device__ inline v8f zero8() {
  v8f z = {0.f, 0.f, 0.f, 0.f, 0.f, 0.f, 0.f, 0.f};
  return z;
}

__device__ inline v16bf cat16(v8bf lo, v8bf hi) {
  return __builtin_shufflevector(lo, hi, 0,1,2,3,4,5,6,7,8,9,10,11,12,13,14,15);
}

// A fragment (16x32 bf16): lane L holds row M=L&15; contiguous K-run
// 2*koff+0..15 with koff = 8*(L>>4). Two 16B loads.
__device__ inline v16bf load_frag_a(const bf16* row, int koff) {
  v8bf lo = *(const v8bf*)(row + 2 * koff);
  v8bf hi = *(const v8bf*)(row + 2 * koff + 8);
  return cat16(lo, hi);
}

// B fragment (32x16 bf16): lane L holds col N=L&15; contiguous K-run of 16
// starting at 16*(L>>4). One 32B (two 16B) load from a K-major row.
__device__ inline v16bf load_frag_b(const bf16* p) {
  v8bf lo = *(const v8bf*)(p);
  v8bf hi = *(const v8bf*)(p + 8);
  return cat16(lo, hi);
}

__device__ inline v8f wmma_bf16(v16bf a, v16bf b, v8f c) {
  // (neg_a, A, neg_b, B, c_mod, C, reuse_a, reuse_b)
  return __builtin_amdgcn_wmma_f32_16x16x32_bf16(false, a, false, b, (short)0, c,
                                                 false, false);
}

__device__ inline float gelu_f(float x) {
  const float k = 0.7978845608028654f; // sqrt(2/pi)
  float x3 = x * x * x;
  return 0.5f * x * (1.0f + tanhf(k * (x + 0.044715f * x3)));
}

// ---------------------------------------------------------------------------
// Tensor Data Mover (TDM) helpers: 2-D bf16 tile load, global -> LDS, with
// hardware row padding. Descriptor layout per CDNA5 ISA 08_async_tensor §8.
// Tile is fixed 64 (K, contiguous) x 128 (rows); LDS rows padded to 72 elems
// (pad 16B after every 128B stored) so fragment ds_load_b128s are
// bank-conflict-free. All fields are workgroup-uniform -> SGPR descriptor.
// This toolchain exposes the 6-arg builtin:
//   (uint32x4 g0, int32x8 g1, int32x4 g2, int32x4 g3, int32x8, i32 cpol)
// ---------------------------------------------------------------------------
__device__ inline v8i tdm_make_g1(int tensor_d0, int tensor_d1, long long stride0) {
  v8i g;
  // [15:0] wg_mask=0 | [17:16] data_size=1 (2B) | [20] pad_enable
  // [24:22] pad_interval=4 (128B) | [31:25] pad_amount=3 (4 DWORDs = 16B)
  g[0] = (1 << 16) | (1 << 20) | (4 << 22) | (3 << 25);
  g[1] = (tensor_d0 & 0xFFFF) << 16;                         // td0[15:0] @ bits[63:48]
  g[2] = (int)(((u32)tensor_d0 >> 16) | ((u32)(tensor_d1 & 0xFFFF) << 16));
  g[3] = (int)(((u32)tensor_d1 >> 16) | (64u << 16));        // tile_dim0 = 64
  g[4] = 128;                                                 // tile_dim1 = 128, tile_dim2 = 0
  g[5] = (int)(stride0 & 0xFFFFFFFFll);                      // dim0 stride [31:0]
  g[6] = (int)((stride0 >> 32) & 0xFFFFll);                  // dim0 stride [47:32]
  g[7] = 0;
  return g;
}

__device__ inline void tdm_load_2d(u32 lds_addr, const bf16* gptr, v8i g1) {
  unsigned long long ga = (unsigned long long)gptr;
  v4u g0;
  g0[0] = 1u;                                // count=1, user descriptor
  g0[1] = lds_addr;                          // LDS byte address
  g0[2] = (u32)ga;                           // global_addr[31:0]
  g0[3] = (u32)(ga >> 32) | (2u << 30);      // global_addr[56:32] | type=2
  v4i z4 = {0, 0, 0, 0};
  v8i z8 = {0, 0, 0, 0, 0, 0, 0, 0};
  __builtin_amdgcn_tensor_load_to_lds(g0, g1, z4, z4, z8, 0);
}

__device__ inline u32 lds_addr_of(const void* p) {
  return (u32)(unsigned long long)p;         // low 32 bits = LDS byte offset
}

// ---------------------------------------------------------------------------
// Weight convert + transpose: fp32 [K][N] -> bf16 [N][K] (so WMMA B-fragments
// are contiguous 32B runs along K). Classic LDS tile transpose.
// ---------------------------------------------------------------------------
__global__ __launch_bounds__(256) void wt_cvt_kernel(const float* __restrict__ w,
                                                     bf16* __restrict__ wt,
                                                     int K, int N) {
  __shared__ float tile[32][33];
  const int n0 = blockIdx.x * 32, k0 = blockIdx.y * 32;
  const int tx = threadIdx.x, ty = threadIdx.y;
  #pragma unroll
  for (int i = ty; i < 32; i += 8)
    tile[i][tx] = w[(size_t)(k0 + i) * N + n0 + tx];
  __syncthreads();
  #pragma unroll
  for (int i = ty; i < 32; i += 8)
    wt[(size_t)(n0 + i) * K + k0 + tx] = (bf16)tile[tx][i];
}

// ---------------------------------------------------------------------------
// LayerNorm (fp32 in -> bf16 out), one row (C=2048) per 256-thread block.
// ---------------------------------------------------------------------------
__global__ __launch_bounds__(256) void layernorm_kernel(const float* __restrict__ x,
                                                        const float* __restrict__ sc,
                                                        const float* __restrict__ bi,
                                                        bf16* __restrict__ h, int Cd) {
  __shared__ float rs[256], rss[256];
  const int row = blockIdx.x, tid = threadIdx.x;
  const float* xr = x + (size_t)row * Cd;
  float s = 0.f, ss = 0.f;
  for (int i = tid; i < Cd; i += 256) { float v = xr[i]; s += v; ss += v * v; }
  rs[tid] = s; rss[tid] = ss;
  __syncthreads();
  for (int st = 128; st > 0; st >>= 1) {
    if (tid < st) { rs[tid] += rs[tid + st]; rss[tid] += rss[tid + st]; }
    __syncthreads();
  }
  const float mu = rs[0] / Cd;
  const float var = rss[0] / Cd - mu * mu;
  const float rstd = rsqrtf(var + 1e-6f);
  bf16* hr = h + (size_t)row * Cd;
  for (int i = tid; i < Cd; i += 256)
    hr[i] = (bf16)((xr[i] - mu) * rstd * sc[i] + bi[i]);
}

// ---------------------------------------------------------------------------
// bf16 WMMA GEMM with TDM double-buffered LDS tiles.
// out = A[MxK] @ Bt^T (Bt stored [N][K]) + bias, optional GELU,
// optional fp32 residual, fp32 and/or bf16 outputs.
// Block tile 128x128, K-step 64; 8 waves, wave tile 32x64 (2x4 WMMA tiles,
// 16 WMMA per K-step). Wave 0 drives the TDM pipeline (EXEC-independent,
// TENSORcnt-tracked); __syncthreads publishes tiles to all waves.
// LDS rows padded to 72 bf16 (144B, inserted by TDM pad hardware) ->
// conflict-free, 16B-aligned ds_load_b128 fragment reads.
// ---------------------------------------------------------------------------
__global__ __launch_bounds__(256) void gemm_bf16_kernel(
    const bf16* __restrict__ A, int lda,
    const bf16* __restrict__ Bt, int ldb,
    const float* __restrict__ bias,
    const float* __restrict__ residual,
    float* __restrict__ out_f32,
    bf16* __restrict__ out_bf,
    int M, int N, int K, int do_gelu) {
  __shared__ alignas(16) bf16 Asm[2][128][72];
  __shared__ alignas(16) bf16 Bsm[2][128][72];

  const int tid = threadIdx.x;
  const int wave = tid >> 5, lane = tid & 31;
  const int half = lane >> 4, ln = lane & 15;
  const int wm = wave >> 1, wn = wave & 1;
  const int m_base = blockIdx.y * 128, n_base = blockIdx.x * 128;
  const int koff = half * 8;

  v8f acc[2][4];
  #pragma unroll
  for (int mi = 0; mi < 2; ++mi)
    #pragma unroll
    for (int ni = 0; ni < 4; ++ni) acc[mi][ni] = zero8();

  const bf16* a_tile = A  + (size_t)m_base * lda;
  const bf16* b_tile = Bt + (size_t)n_base * ldb;
  const int nk = K >> 6;

  if (wave == 0) {
    const v8i g1a = tdm_make_g1(K, M, lda);
    const v8i g1b = tdm_make_g1(K, N, ldb);
    tdm_load_2d(lds_addr_of(&Asm[0][0][0]), a_tile, g1a);
    tdm_load_2d(lds_addr_of(&Bsm[0][0][0]), b_tile, g1b);
    for (int ki = 0; ki < nk; ++ki) {
      if (ki + 1 < nk) {
        const int nxt = (ki + 1) & 1;
        const int k1 = (ki + 1) << 6;
        tdm_load_2d(lds_addr_of(&Asm[nxt][0][0]), a_tile + k1, g1a);
        tdm_load_2d(lds_addr_of(&Bsm[nxt][0][0]), b_tile + k1, g1b);
        __builtin_amdgcn_s_wait_tensorcnt(2);  // tile ki resident, ki+1 streaming
      } else {
        __builtin_amdgcn_s_wait_tensorcnt(0);
      }
      __syncthreads();  // publish tile ki
      const int cur = ki & 1;
      #pragma unroll
      for (int ks = 0; ks < 2; ++ks) {
        v16bf af[2], bfrag[4];
        #pragma unroll
        for (int mi = 0; mi < 2; ++mi)
          af[mi] = load_frag_a(&Asm[cur][wm * 32 + mi * 16 + ln][ks * 32], koff);
        #pragma unroll
        for (int ni = 0; ni < 4; ++ni)
          bfrag[ni] = load_frag_b(&Bsm[cur][wn * 64 + ni * 16 + ln][ks * 32 + half * 16]);
        #pragma unroll
        for (int mi = 0; mi < 2; ++mi)
          #pragma unroll
          for (int ni = 0; ni < 4; ++ni)
            acc[mi][ni] = wmma_bf16(af[mi], bfrag[ni], acc[mi][ni]);
      }
      __syncthreads();  // all waves done reading before buffer reuse
    }
  } else {
    for (int ki = 0; ki < nk; ++ki) {
      __syncthreads();
      const int cur = ki & 1;
      #pragma unroll
      for (int ks = 0; ks < 2; ++ks) {
        v16bf af[2], bfrag[4];
        #pragma unroll
        for (int mi = 0; mi < 2; ++mi)
          af[mi] = load_frag_a(&Asm[cur][wm * 32 + mi * 16 + ln][ks * 32], koff);
        #pragma unroll
        for (int ni = 0; ni < 4; ++ni)
          bfrag[ni] = load_frag_b(&Bsm[cur][wn * 64 + ni * 16 + ln][ks * 32 + half * 16]);
        #pragma unroll
        for (int mi = 0; mi < 2; ++mi)
          #pragma unroll
          for (int ni = 0; ni < 4; ++ni)
            acc[mi][ni] = wmma_bf16(af[mi], bfrag[ni], acc[mi][ni]);
      }
      __syncthreads();
    }
  }

  // Epilogue: C layout element r -> row = r + 8*half, col = ln (within tile).
  #pragma unroll
  for (int mi = 0; mi < 2; ++mi) {
    const int r0 = m_base + wm * 32 + mi * 16 + 8 * half;
    #pragma unroll
    for (int ni = 0; ni < 4; ++ni) {
      const int col = n_base + wn * 64 + ni * 16 + ln;
      const float bv = bias[col];
      #pragma unroll
      for (int r = 0; r < 8; ++r) {
        const size_t idx = (size_t)(r0 + r) * N + col;
        float v = acc[mi][ni][r] + bv;
        if (do_gelu) v = gelu_f(v);
        if (residual) v += residual[idx];
        if (out_f32) out_f32[idx] = v;
        if (out_bf)  out_bf[idx]  = (bf16)v;
      }
    }
  }
}

// ---------------------------------------------------------------------------
// V transpose: qkv bf16 [B*T][3C] (v at col 2C + h*128) -> vt [B,H,HD,T]
// so AV B-fragments (fixed d, contiguous kv) are contiguous 32B loads.
// ---------------------------------------------------------------------------
__global__ __launch_bounds__(256) void vt_transpose_kernel(const bf16* __restrict__ qkv,
                                                           bf16* __restrict__ vt) {
  const int T = 2048, C = 2048, HD = 128, LDQ = 3 * C, H = 16;
  __shared__ bf16 tile[32][34];
  const int bh = blockIdx.z, b = bh / H, h = bh % H;
  const int t0 = blockIdx.x * 32, d0 = blockIdx.y * 32;
  const int tx = threadIdx.x, ty = threadIdx.y;
  #pragma unroll
  for (int i = ty; i < 32; i += 8)
    tile[i][tx] = qkv[((size_t)b * T + t0 + i) * LDQ + 2 * C + h * HD + d0 + tx];
  __syncthreads();
  #pragma unroll
  for (int i = ty; i < 32; i += 8)
    vt[((size_t)bh * HD + d0 + i) * T + t0 + tx] = tile[tx][i];
}

// ---------------------------------------------------------------------------
// Flash-style causal attention. Block = 128 threads (4 waves); each wave owns
// a 16-row Q tile (fragments resident in VGPRs) and streams 32-wide KV blocks.
// Per KV block: 8 WMMA for S = Q K^T, fp32 online softmax with 16-lane
// shuffle reductions, P -> 1KB per-wave LDS bounce to A-layout, 8 WMMA for AV.
// ---------------------------------------------------------------------------
__global__ __launch_bounds__(128) void attention_kernel(const bf16* __restrict__ qkv,
                                                        const bf16* __restrict__ vt,
                                                        bf16* __restrict__ y) {
  const int T = 2048, C = 2048, HD = 128, LDQ = 3 * C;
  __shared__ alignas(16) bf16 p_lds[4][16][40];

  const int bh = blockIdx.y, b = bh >> 4, h = bh & 15;
  const int wave = threadIdx.x >> 5, lane = threadIdx.x & 31;
  const int half = lane >> 4, ln = lane & 15;
  const int q0 = blockIdx.x * 64 + wave * 16;
  const int koff = half * 8;

  const bf16* qp  = qkv + (size_t)b * T * LDQ + h * HD;  // q cols [0,C)
  const bf16* kp  = qp + C;                              // k cols [C,2C)
  const bf16* vtp = vt + (size_t)bh * HD * T;            // [HD][T]

  // Q fragments for all 4 d-steps (d = s*32), resident across the KV loop.
  v16bf qa[4];
  const bf16* qrow = qp + (size_t)(q0 + ln) * LDQ;
  #pragma unroll
  for (int s = 0; s < 4; ++s) qa[s] = load_frag_a(qrow + s * 32, koff);

  float mr[8], lr[8];
  v8f o[8];
  #pragma unroll
  for (int r = 0; r < 8; ++r) { mr[r] = -1e30f; lr[r] = 0.f; }
  #pragma unroll
  for (int n = 0; n < 8; ++n) o[n] = zero8();

  const float scl = 0.08838834764831845f;  // 1/sqrt(128)
  const int jmax = (q0 + 15) >> 5;

  for (int j = 0; j <= jmax; ++j) {
    const int kv0 = j * 32;
    v8f s0 = zero8(), s1 = zero8();
    #pragma unroll
    for (int s = 0; s < 4; ++s) {
      // B-fragment of K^T: row kv = kv0 + ln, contiguous d-run of 16.
      const bf16* kb = kp + (size_t)(kv0 + ln) * LDQ + s * 32 + half * 16;
      s0 = wmma_bf16(qa[s], load_frag_b(kb), s0);
      s1 = wmma_bf16(qa[s], load_frag_b(kb + (size_t)16 * LDQ), s1);
    }

    const bool need_mask = (kv0 + 31 > q0);
    #pragma unroll
    for (int r = 0; r < 8; ++r) {
      float a0 = s0[r] * scl, a1 = s1[r] * scl;
      if (need_mask) {
        const int q = q0 + r + 8 * half;
        if (kv0 + ln > q)      a0 = -1e30f;
        if (kv0 + 16 + ln > q) a1 = -1e30f;
      }
      float mx = fmaxf(a0, a1);
      mx = fmaxf(mx, __shfl_xor(mx, 1));
      mx = fmaxf(mx, __shfl_xor(mx, 2));
      mx = fmaxf(mx, __shfl_xor(mx, 4));
      mx = fmaxf(mx, __shfl_xor(mx, 8));
      const float mnew  = fmaxf(mr[r], mx);
      const float alpha = __expf(mr[r] - mnew);
      mr[r] = mnew;
      const float p0 = __expf(a0 - mnew);
      const float p1 = __expf(a1 - mnew);
      float rsum = p0 + p1;
      rsum += __shfl_xor(rsum, 1);
      rsum += __shfl_xor(rsum, 2);
      rsum += __shfl_xor(rsum, 4);
      rsum += __shfl_xor(rsum, 8);
      lr[r] = lr[r] * alpha + rsum;
      #pragma unroll
      for (int n = 0; n < 8; ++n) o[n][r] *= alpha;
      // C layout -> LDS tile [q_in_tile][kv_in_block] (row padded to 40).
      p_lds[wave][r + 8 * half][ln]      = (bf16)p0;
      p_lds[wave][r + 8 * half][16 + ln] = (bf16)p1;
    }

    // Re-read P as a WMMA A-fragment (16 q x 32 kv); same-wave LDS is ordered.
    v16bf pa = load_frag_a(&p_lds[wave][ln][0], koff);
    #pragma unroll
    for (int n = 0; n < 8; ++n) {
      // B-fragment of V: vt row d = n*16 + ln, contiguous kv-run of 16.
      const bf16* vb = vtp + (size_t)(n * 16 + ln) * T + kv0 + half * 16;
      o[n] = wmma_bf16(pa, load_frag_b(vb), o[n]);
    }
  }

  #pragma unroll
  for (int n = 0; n < 8; ++n) {
    #pragma unroll
    for (int r = 0; r < 8; ++r) {
      const int t = q0 + r + 8 * half;
      const int c = h * HD + n * 16 + ln;
      y[((size_t)b * T + t) * C + c] = (bf16)(o[n][r] / lr[r]);
    }
  }
}

// ---------------------------------------------------------------------------
// Host-side orchestration.
// ---------------------------------------------------------------------------
extern "C" void kernel_launch(void* const* d_in, const int* in_sizes, int n_in,
                              void* d_out, int out_size, void* d_ws, size_t ws_size,
                              hipStream_t stream) {
  (void)in_sizes; (void)n_in; (void)out_size; (void)ws_size;
  const float* x      = (const float*)d_in[0];
  const float* ln_s   = (const float*)d_in[1];
  const float* ln_b   = (const float*)d_in[2];
  const float* w_qkv  = (const float*)d_in[3];
  const float* b_qkv  = (const float*)d_in[4];
  const float* w_proj = (const float*)d_in[5];
  const float* b_proj = (const float*)d_in[6];
  const float* w_fc   = (const float*)d_in[7];
  const float* b_fc   = (const float*)d_in[8];
  const float* w_out  = (const float*)d_in[9];
  const float* b_out  = (const float*)d_in[10];
  float* out = (float*)d_out;

  const int B = 2, T = 2048, C = 2048, H = 16, HD = 128;
  const int M = B * T;          // 4096 rows
  (void)H; (void)HD;

  // Workspace carve (~304 MB total).
  char* ws = (char*)d_ws;
  size_t off = 0;
  auto carve = [&](size_t bytes) -> void* {
    void* p = ws + off;
    off += (bytes + 255) & ~(size_t)255;
    return p;
  };
  bf16* h_bf   = (bf16*)carve((size_t)M * C * 2);            // 16 MB
  bf16* wqkvT  = (bf16*)carve((size_t)3 * C * C * 2);        // 24 MB [6144][2048]
  bf16* wprojT = (bf16*)carve((size_t)C * C * 2);            //  8 MB [2048][2048]
  bf16* wfcT   = (bf16*)carve((size_t)4 * C * C * 2);        // 32 MB [8192][2048]
  bf16* woutT  = (bf16*)carve((size_t)4 * C * C * 2);        // 32 MB [2048][8192]
  bf16* qkv_bf = (bf16*)carve((size_t)M * 3 * C * 2);        // 48 MB [4096][6144]
  bf16* vtb    = (bf16*)carve((size_t)B * H * HD * T * 2);   // 16 MB [B,H,HD,T]
  bf16* y_bf   = (bf16*)carve((size_t)M * C * 2);            // 16 MB
  float* x1    = (float*)carve((size_t)M * C * 4);           // 32 MB
  bf16* x1_bf  = (bf16*)carve((size_t)M * C * 2);            // 16 MB
  bf16* m_bf   = (bf16*)carve((size_t)M * 4 * C * 2);        // 64 MB

  const dim3 t32x8(32, 8);

  // 1) Weights -> bf16, transposed to [N][K].
  wt_cvt_kernel<<<dim3(6144 / 32, 2048 / 32), t32x8, 0, stream>>>(w_qkv,  wqkvT,  2048, 6144);
  wt_cvt_kernel<<<dim3(2048 / 32, 2048 / 32), t32x8, 0, stream>>>(w_proj, wprojT, 2048, 2048);
  wt_cvt_kernel<<<dim3(8192 / 32, 2048 / 32), t32x8, 0, stream>>>(w_fc,   wfcT,   2048, 8192);
  wt_cvt_kernel<<<dim3(2048 / 32, 8192 / 32), t32x8, 0, stream>>>(w_out,  woutT,  8192, 2048);

  // 2) LayerNorm -> bf16 h.
  layernorm_kernel<<<M, 256, 0, stream>>>(x, ln_s, ln_b, h_bf, C);

  // 3) QKV GEMM: [4096,2048] x [2048,6144] + bias -> bf16 qkv.
  gemm_bf16_kernel<<<dim3(6144 / 128, M / 128), 256, 0, stream>>>(
      h_bf, C, wqkvT, C, b_qkv, nullptr, nullptr, qkv_bf, M, 3 * C, C, 0);

  // 4) V -> [B,H,HD,T] for contiguous AV B-fragments.
  vt_transpose_kernel<<<dim3(T / 32, HD / 32, B * H), t32x8, 0, stream>>>(qkv_bf, vtb);

  // 5) Flash attention -> bf16 y [B,T,C].
  attention_kernel<<<dim3(T / 64, B * H), 128, 0, stream>>>(qkv_bf, vtb, y_bf);

  // 6) Proj GEMM + residual x: -> x1 (fp32) and x1_bf (bf16).
  gemm_bf16_kernel<<<dim3(2048 / 128, M / 128), 256, 0, stream>>>(
      y_bf, C, wprojT, C, b_proj, x, x1, x1_bf, M, C, C, 0);

  // 7) FC GEMM + GELU -> bf16 m [4096,8192].
  gemm_bf16_kernel<<<dim3(8192 / 128, M / 128), 256, 0, stream>>>(
      x1_bf, C, wfcT, C, b_fc, nullptr, nullptr, m_bf, M, 4 * C, C, 1);

  // 8) Out GEMM + residual x1 -> d_out (fp32).
  gemm_bf16_kernel<<<dim3(2048 / 128, M / 128), 256, 0, stream>>>(
      m_bf, 4 * C, woutT, 4 * C, b_out, x1, out, nullptr, M, C, 4 * C, 0);
}
